// WanSelfAttention_46273977647626
// MI455X (gfx1250) — matrix operand order repaired and checked by reading, compiled, not run on hardware
//
#include <hip/hip_runtime.h>
#include <hip/hip_bf16.h>

// ---------------------------------------------------------------------------
// CDNA5 (gfx1250) WMMA / TDM types
// ---------------------------------------------------------------------------
typedef __attribute__((ext_vector_type(16))) __bf16    v16bf;
typedef __attribute__((ext_vector_type(8)))  float     v8f;
typedef __attribute__((ext_vector_type(4)))  unsigned  u32x4;
typedef __attribute__((ext_vector_type(8)))  unsigned  u32x8;

union BF16x16 {            // 32 bytes: one A-frag / B-frag register image
    uint4 u[2];
    v16bf v;
};

static __device__ __forceinline__ unsigned short f2bf(float f) {
    unsigned int u = __float_as_uint(f);
    unsigned int lsb = (u >> 16) & 1u;
    u += 0x7fffu + lsb;                   // round to nearest even
    return (unsigned short)(u >> 16);
}

// ---------------------------------------------------------------------------
// Tensor Data Mover: 2D tile (bf16) global -> LDS, with HW row padding.
// flags = D# group1 dword0 (data_size=2B, pad_enable, pad_interval/amount).
// ---------------------------------------------------------------------------
static __device__ __forceinline__ void tdm_load_2d(
    unsigned lds_addr, unsigned long long gaddr, unsigned flags,
    unsigned td0, unsigned td1, unsigned tile0, unsigned tile1, unsigned stride0)
{
    u32x4 g0 = { 1u,                                   // count=1, user desc
                 lds_addr,                             // lds_addr [63:32]
                 (unsigned)gaddr,                      // global_addr [95:64]
                 (unsigned)((gaddr >> 32) & 0x01FFFFFFu) | 0x80000000u }; // type=2
    u32x8 g1 = { flags,
                 (td0 & 0xFFFFu) << 16,                               // td0 lo
                 ((td0 >> 16) & 0xFFFFu) | ((td1 & 0xFFFFu) << 16),   // td0 hi | td1 lo
                 ((td1 >> 16) & 0xFFFFu) | ((tile0 & 0xFFFFu) << 16), // td1 hi | tile0
                 (tile1 & 0xFFFFu),                                   // tile1 | tile2=0
                 stride0,                                             // stride0 lo
                 0u, 0u };
    asm volatile("tensor_load_to_lds %0, %1" : : "s"(g0), "s"(g1) : "memory");
}

// flag words: data_size=1(2B)<<16 | pad_en<<20 | interval<<22 | amount(4DW)=3<<25
#define TDM_FLAGS_ROW64B   0x6D10000u   // interval 16 DW -> pad to 40-short rows
#define TDM_FLAGS_ROW128B  0x7110000u   // interval 32 DW -> pad to 72-short rows
#define TDM_FLAGS_ROW256B  0x7510000u   // interval 64 DW -> pad to 136-short rows

// ---------------------------------------------------------------------------
// Pack kernels: f32 -> bf16 (row-major) and f32 [N][K] -> bf16 [K][N]
// ---------------------------------------------------------------------------
__global__ __launch_bounds__(256) void pack_bf16_kernel(
    const float* __restrict__ src, unsigned short* __restrict__ dst, int n4)
{
    int i = blockIdx.x * 256 + threadIdx.x;
    if (i >= n4) return;
    float4 f = *(const float4*)(src + (size_t)i * 4);
    uint2 o;
    o.x = f2bf(f.x) | ((unsigned)f2bf(f.y) << 16);
    o.y = f2bf(f.z) | ((unsigned)f2bf(f.w) << 16);
    *(uint2*)(dst + (size_t)i * 4) = o;
}

__global__ __launch_bounds__(256) void pack_wt_kernel(
    const float* __restrict__ W, unsigned short* __restrict__ Wt)
{
    __shared__ unsigned short T[32][33];
    int n0 = blockIdx.x * 32, k0 = blockIdx.y * 32;
    int t = threadIdx.x;
    int r = t >> 3, c4 = (t & 7) << 2;
    float4 f = *(const float4*)(W + (size_t)(n0 + r) * 1536 + k0 + c4);
    T[c4 + 0][r] = f2bf(f.x);
    T[c4 + 1][r] = f2bf(f.y);
    T[c4 + 2][r] = f2bf(f.z);
    T[c4 + 3][r] = f2bf(f.w);
    __syncthreads();
    const unsigned short* p = &T[r][c4];
    uint2 o;
    o.x = p[0] | ((unsigned)p[1] << 16);
    o.y = p[2] | ((unsigned)p[3] << 16);
    *(uint2*)(Wt + (size_t)(k0 + r) * 1536 + n0 + c4) = o;
}

// ---------------------------------------------------------------------------
// GEMM: C[2400 x 1536] f32 = A[2400 x 1536](bf16) * Wt[1536 x 1536](bf16) + bias
// Wt is pre-transposed: Wt[k][n].  Block tile 96x128, K-step 32, 8 waves (2x4).
// Tiles move via TDM, double buffered; LDS rows padded by TDM pad feature.
// ---------------------------------------------------------------------------
#define GK_BM     96
#define GK_BN     128
#define A_STRIDE  40      // shorts per LDS A row (64B data + 16B pad)
#define B_STRIDE  136     // shorts per LDS B row (256B data + 16B pad)

__global__ __launch_bounds__(256) void gemm_bias_kernel(
    const unsigned short* __restrict__ A,
    const unsigned short* __restrict__ Wt,
    const float* __restrict__ bias,
    float* __restrict__ C)
{
    __shared__ __align__(16) unsigned short As[2][GK_BM * A_STRIDE];
    __shared__ __align__(16) unsigned short Bs[2][32 * B_STRIDE];

    const int m0 = blockIdx.x * GK_BM;
    const int n0 = blockIdx.y * GK_BN;
    const int t = threadIdx.x;
    const int lane = t & 31;
    const int wid = t >> 5;
    const int wm = wid >> 2, wn = wid & 3;
    const int mlo = lane & 15, lg = lane >> 4;

    v8f acc[3][2];
#pragma unroll
    for (int i = 0; i < 3; i++)
#pragma unroll
        for (int j = 0; j < 2; j++) acc[i][j] = {};

    if (wid == 0) {
        tdm_load_2d((unsigned)(size_t)&As[0][0],
                    (unsigned long long)A + ((unsigned long long)m0 * 1536) * 2,
                    TDM_FLAGS_ROW64B, 1536, 96, 32, 96, 1536);
        tdm_load_2d((unsigned)(size_t)&Bs[0][0],
                    (unsigned long long)Wt + ((unsigned long long)n0) * 2,
                    TDM_FLAGS_ROW256B, 1536 - n0, 32, 128, 32, 1536);
    }

    for (int i = 0; i < 48; i++) {
        if (wid == 0) {
            if (i + 1 < 48) {
                int k0 = (i + 1) * 32;
                tdm_load_2d((unsigned)(size_t)&As[(i + 1) & 1][0],
                            (unsigned long long)A +
                                ((unsigned long long)m0 * 1536 + k0) * 2,
                            TDM_FLAGS_ROW64B, 1536 - k0, 96, 32, 96, 1536);
                tdm_load_2d((unsigned)(size_t)&Bs[(i + 1) & 1][0],
                            (unsigned long long)Wt +
                                ((unsigned long long)k0 * 1536 + n0) * 2,
                            TDM_FLAGS_ROW256B, 1536 - n0, 32, 128, 32, 1536);
                asm volatile("s_wait_tensorcnt 0x2" ::: "memory");
            } else {
                asm volatile("s_wait_tensorcnt 0x0" ::: "memory");
            }
        }
        __syncthreads();

        const unsigned short* Asb = As[i & 1];
        const unsigned short* Bsb = Bs[i & 1];
        BF16x16 afr[3], bfr[2];
#pragma unroll
        for (int im = 0; im < 3; im++) {
            const unsigned short* p =
                &Asb[(wm * 48 + im * 16 + mlo) * A_STRIDE + lg * 8];
            afr[im].u[0] = *(const uint4*)p;         // K = lg*8 .. +7
            afr[im].u[1] = *(const uint4*)(p + 16);  // K = 16+lg*8 .. +7
        }
#pragma unroll
        for (int in = 0; in < 2; in++) {
            const unsigned short* p = &Bsb[wn * 32 + in * 16 + mlo];
#pragma unroll
            for (int s = 0; s < 16; s++)
                ((unsigned short*)&bfr[in])[s] = p[(((s & 8) << 1) + lg * 8 + (s & 7)) * B_STRIDE];
        }
#pragma unroll
        for (int im = 0; im < 3; im++)
#pragma unroll
            for (int in = 0; in < 2; in++)
                acc[im][in] = __builtin_amdgcn_wmma_f32_16x16x32_bf16(
                    false, afr[im].v, false, bfr[in].v,
                    (short)0, acc[im][in], false, false);
        __syncthreads();
    }

#pragma unroll
    for (int im = 0; im < 3; im++)
#pragma unroll
        for (int in = 0; in < 2; in++) {
            int col = n0 + wn * 32 + in * 16 + mlo;
            float b = bias[col];
#pragma unroll
            for (int r = 0; r < 8; r++) {
                int row = m0 + wm * 48 + im * 16 + r + 8 * lg;
                C[(size_t)row * 1536 + col] = acc[im][in][r] + b;
            }
        }
}

// ---------------------------------------------------------------------------
// RMSNorm(q,k) + RoPE + bf16 packing (q row-major, k transposed [hd][L], v)
// ---------------------------------------------------------------------------
__global__ __launch_bounds__(256) void norm_rope_pack_kernel(
    const float* __restrict__ qf, const float* __restrict__ kf,
    const float* __restrict__ vf,
    const float* __restrict__ gq, const float* __restrict__ gk,
    const float* __restrict__ freqs,
    unsigned short* __restrict__ qb,         // [2400][1536]
    unsigned short* __restrict__ ktb,        // [1536][2400] (hd-major)
    unsigned short* __restrict__ vb)         // [2400][1536]
{
    const int L = 2400, D = 1536;
    int row = blockIdx.x;
    int t = threadIdx.x;
    __shared__ float red[256];
    __shared__ float qn[1536], kn[1536];

    const float* qr = qf + (size_t)row * D;
    const float* kr = kf + (size_t)row * D;

    float sq = 0.f, sk = 0.f;
    for (int i = t; i < D; i += 256) {
        float a = qr[i]; sq += a * a;
        float b = kr[i]; sk += b * b;
    }
    red[t] = sq; __syncthreads();
    for (int s = 128; s > 0; s >>= 1) { if (t < s) red[t] += red[t + s]; __syncthreads(); }
    float rq = rsqrtf(red[0] / (float)D + 1e-6f);
    __syncthreads();
    red[t] = sk; __syncthreads();
    for (int s = 128; s > 0; s >>= 1) { if (t < s) red[t] += red[t + s]; __syncthreads(); }
    float rk = rsqrtf(red[0] / (float)D + 1e-6f);
    __syncthreads();

    for (int i = t; i < D; i += 256) {
        qn[i] = qr[i] * rq * gq[i];
        kn[i] = kr[i] * rk * gk[i];
    }
    __syncthreads();

    int fi = row / 400, hi = (row / 20) % 20, wi = row % 20;
    for (int p = t; p < 768; p += 256) {
        int head = p >> 6, c = p & 63;
        int pos = (c < 22) ? fi : ((c < 43) ? hi : wi);
        const float* R = freqs + ((size_t)pos * 64 + c) * 4;
        float r00 = R[0], r01 = R[1], r10 = R[2], r11 = R[3];
        int d0 = head * 128 + 2 * c;
        float q0 = qn[d0], q1 = qn[d0 + 1];
        float k0 = kn[d0], k1 = kn[d0 + 1];
        size_t o = (size_t)row * D + d0;
        qb[o]     = f2bf(r00 * q0 + r01 * q1);
        qb[o + 1] = f2bf(r10 * q0 + r11 * q1);
        ktb[(size_t)(d0)     * L + row] = f2bf(r00 * k0 + r01 * k1);
        ktb[(size_t)(d0 + 1) * L + row] = f2bf(r10 * k0 + r11 * k1);
    }
    const float* vr = vf + (size_t)row * D;
    for (int i = t; i < D; i += 256) vb[(size_t)row * D + i] = f2bf(vr[i]);
}

// ---------------------------------------------------------------------------
// Flash attention. grid=(19, 12 heads), block=256 (8 waves x 16 q-rows).
// Per key-chunk (64): K tile staged by TDM (wave 0), V tile staged by per-lane
// async-to-LDS loads (all waves); all 8 waves consume from LDS.
// ---------------------------------------------------------------------------
__global__ __launch_bounds__(256) void attn_kernel(
    const unsigned short* __restrict__ qb,   // [2400][1536]
    const unsigned short* __restrict__ ktb,  // [1536][2400]
    const unsigned short* __restrict__ vb,   // [2400][1536]
    unsigned short* __restrict__ outb)       // [2400][1536] bf16
{
    const int L = 2400, D = 1536;
    const int h = blockIdx.y;
    const int t = threadIdx.x, lane = t & 31, wid = t >> 5;
    const int q0 = blockIdx.x * 128 + wid * 16;
    const bool active = (q0 < L);

    __shared__ __align__(16) unsigned short Ks[128 * 72];   // K tile [hd][key]
    __shared__ __align__(16) unsigned short Vs[64 * 136];   // V tile [key][hd]
    __shared__ __align__(16) float          Sld[8][16 * 64];
    __shared__ __align__(16) unsigned short Pld[8][16 * 64];
    __shared__ float Corr[8][16];
    __shared__ float Lsum[8][16];

    const int mlo = lane & 15, lg = lane >> 4;

    // Q fragments: 16 rows x 128 hd -> four 16x32 A-frags (clamped row if OOB)
    BF16x16 qfr[4];
    {
        int qrow = q0 + mlo; if (qrow >= L) qrow = L - 1;
        const unsigned short* p = qb + (size_t)qrow * D + h * 128;
#pragma unroll
        for (int kc = 0; kc < 4; kc++) {
            qfr[kc].u[0] = *(const uint4*)(p + kc * 32 + lg * 8);
            qfr[kc].u[1] = *(const uint4*)(p + kc * 32 + 16 + lg * 8);
        }
    }

    v8f o[8];
#pragma unroll
    for (int i = 0; i < 8; i++) o[i] = {};
    float m_i = -1e30f, l_i = 0.f;

    float*          Sw = Sld[wid];
    unsigned short* Pw = Pld[wid];
    float*          Cw = Corr[wid];
    const float scale = 0.088388347648318447f;   // 1/sqrt(128)
    const unsigned vs_base = (unsigned)(size_t)&Vs[0];

    for (int j = 0; j < 38; j++) {
        int kbase = j * 64;

        // ---- stage K tile via TDM (wave 0): 128 hd-rows x 64 keys ----
        if (wid == 0) {
            tdm_load_2d((unsigned)(size_t)&Ks[0],
                        (unsigned long long)ktb +
                            ((unsigned long long)(h * 128) * L + kbase) * 2,
                        TDM_FLAGS_ROW128B, (unsigned)(L - kbase), 128, 64, 128,
                        (unsigned)L);
        }
        // ---- stage V tile via async-to-LDS: 64 key-rows x 128 hd ----
#pragma unroll
        for (int pcs = 0; pcs < 4; pcs++) {
            int row = pcs * 16 + (t >> 4);       // 0..63
            int c8  = t & 15;                    // 16B column
            int rr = kbase + row; if (rr >= L) rr = L - 1;
            unsigned long long ga = (unsigned long long)vb +
                ((unsigned long long)rr * D + h * 128 + c8 * 8) * 2;
            unsigned la = vs_base + (unsigned)(row * 272 + c8 * 16);
            asm volatile("global_load_async_to_lds_b128 %0, %1, off"
                         : : "v"(la), "v"(ga) : "memory");
        }
        asm volatile("s_wait_asynccnt 0x0" ::: "memory");
        if (wid == 0) asm volatile("s_wait_tensorcnt 0x0" ::: "memory");
        __syncthreads();

        if (active) {
            // ---- S = (Q K^T) * scale ----
#pragma unroll
            for (int nt = 0; nt < 4; nt++) {
                v8f s = {};
#pragma unroll
                for (int kc = 0; kc < 4; kc++) {
                    BF16x16 b;
                    const unsigned short* p = &Ks[(kc * 32) * 72 + nt * 16 + mlo];
#pragma unroll
                    for (int s = 0; s < 16; s++)
                        ((unsigned short*)&b)[s] = p[(((s & 8) << 1) + lg * 8 + (s & 7)) * 72];
                    s = __builtin_amdgcn_wmma_f32_16x16x32_bf16(
                        false, qfr[kc].v, false, b.v, (short)0, s, false, false);
                }
#pragma unroll
                for (int r = 0; r < 8; r++)
                    Sw[(r + 8 * lg) * 64 + nt * 16 + mlo] = s[r] * scale;
            }
            asm volatile("s_wait_dscnt 0x0" ::: "memory");

            // ---- online softmax, one row per lane (lanes 0..15) ----
            if (lane < 16) {
                float cm = -1e30f;
                for (int c = 0; c < 64; c++) {
                    float x = Sw[lane * 64 + c];
                    if (kbase + c >= L) x = -1e9f;
                    cm = fmaxf(cm, x);
                }
                float mnew = fmaxf(m_i, cm);
                float corr = __expf(m_i - mnew);
                float ssum = 0.f;
                for (int c = 0; c < 64; c++) {
                    float x = Sw[lane * 64 + c];
                    if (kbase + c >= L) x = -1e9f;
                    float pv = __expf(x - mnew);
                    ssum += pv;
                    Pw[lane * 64 + c] = f2bf(pv);
                }
                m_i = mnew;
                l_i = l_i * corr + ssum;
                Cw[lane] = corr;
            }
            asm volatile("s_wait_dscnt 0x0" ::: "memory");

            // ---- rescale O ----
            float cr[8];
#pragma unroll
            for (int r = 0; r < 8; r++) cr[r] = Cw[r + 8 * lg];
#pragma unroll
            for (int ht = 0; ht < 8; ht++)
#pragma unroll
                for (int r = 0; r < 8; r++) o[ht][r] *= cr[r];

            // ---- O += P @ V ----
#pragma unroll
            for (int kk = 0; kk < 2; kk++) {
                BF16x16 a;
                const unsigned short* pp = Pw + mlo * 64 + kk * 32;
                a.u[0] = *(const uint4*)(pp + lg * 8);
                a.u[1] = *(const uint4*)(pp + 16 + lg * 8);
#pragma unroll
                for (int ht = 0; ht < 8; ht++) {
                    BF16x16 b;
                    const unsigned short* vp = &Vs[(kk * 32) * 136 + ht * 16 + mlo];
#pragma unroll
                    for (int s = 0; s < 16; s++)
                        ((unsigned short*)&b)[s] = vp[(((s & 8) << 1) + lg * 8 + (s & 7)) * 136];
                    o[ht] = __builtin_amdgcn_wmma_f32_16x16x32_bf16(
                        false, a.v, false, b.v, (short)0, o[ht], false, false);
                }
            }
        }
        __syncthreads();
    }

    if (!active) return;
    if (lane < 16) Lsum[wid][lane] = l_i;
    asm volatile("s_wait_dscnt 0x0" ::: "memory");
    float inv[8];
#pragma unroll
    for (int r = 0; r < 8; r++) inv[r] = 1.f / Lsum[wid][r + 8 * lg];
#pragma unroll
    for (int ht = 0; ht < 8; ht++)
#pragma unroll
        for (int r = 0; r < 8; r++) {
            int row = q0 + r + 8 * lg;
            int col = h * 128 + ht * 16 + mlo;
            outb[(size_t)row * D + col] = f2bf(o[ht][r] * inv[r]);
        }
}

// ---------------------------------------------------------------------------
// Launch
// ---------------------------------------------------------------------------
extern "C" void kernel_launch(void* const* d_in, const int* in_sizes, int n_in,
                              void* d_out, int out_size, void* d_ws, size_t ws_size,
                              hipStream_t stream) {
    (void)in_sizes; (void)n_in; (void)out_size; (void)ws_size;
    const float* x     = (const float*)d_in[0];
    const float* Wq    = (const float*)d_in[1];
    const float* bq    = (const float*)d_in[2];
    const float* Wk    = (const float*)d_in[3];
    const float* bk    = (const float*)d_in[4];
    const float* Wv    = (const float*)d_in[5];
    const float* bv    = (const float*)d_in[6];
    const float* Wo    = (const float*)d_in[7];
    const float* bo    = (const float*)d_in[8];
    const float* gq    = (const float*)d_in[9];
    const float* gk    = (const float*)d_in[10];
    const float* freqs = (const float*)d_in[13];

    const size_t F32MAT  = (size_t)2400 * 1536 * 4;   // 14,745,600 B
    const size_t BF16MAT = (size_t)2400 * 1536 * 2;   //  7,372,800 B
    const size_t BF16W   = (size_t)1536 * 1536 * 2;   //  4,718,592 B
    char* ws = (char*)d_ws;
    unsigned short* xb    = (unsigned short*)ws;  ws += BF16MAT;
    unsigned short* wtq   = (unsigned short*)ws;  ws += BF16W;
    unsigned short* wtk   = (unsigned short*)ws;  ws += BF16W;
    unsigned short* wtv   = (unsigned short*)ws;  ws += BF16W;
    unsigned short* wto   = (unsigned short*)ws;  ws += BF16W;
    float* qf   = (float*)ws;                     ws += F32MAT;
    float* kf   = (float*)ws;                     ws += F32MAT;
    float* vf   = (float*)ws;                     ws += F32MAT;
    unsigned short* qbb   = (unsigned short*)ws;  ws += BF16MAT;
    unsigned short* ktb   = (unsigned short*)ws;  ws += BF16MAT;
    unsigned short* vbb   = (unsigned short*)ws;  ws += BF16MAT;
    unsigned short* attnb = (unsigned short*)ws;  ws += BF16MAT;

    dim3 gg(25, 12), gb(256);
    dim3 gw(48, 48);
    pack_bf16_kernel<<<3600, 256, 0, stream>>>(x, xb, 921600);
    pack_wt_kernel<<<gw, gb, 0, stream>>>(Wq, wtq);
    pack_wt_kernel<<<gw, gb, 0, stream>>>(Wk, wtk);
    pack_wt_kernel<<<gw, gb, 0, stream>>>(Wv, wtv);
    pack_wt_kernel<<<gw, gb, 0, stream>>>(Wo, wto);
    gemm_bias_kernel<<<gg, gb, 0, stream>>>(xb, wtq, bq, qf);
    gemm_bias_kernel<<<gg, gb, 0, stream>>>(xb, wtk, bk, kf);
    gemm_bias_kernel<<<gg, gb, 0, stream>>>(xb, wtv, bv, vf);
    norm_rope_pack_kernel<<<2400, 256, 0, stream>>>(qf, kf, vf, gq, gk, freqs,
                                                    qbb, ktb, vbb);
    attn_kernel<<<dim3(19, 12), 256, 0, stream>>>(qbb, ktb, vbb, attnb);
    gemm_bias_kernel<<<gg, gb, 0, stream>>>(attnb, wto, bo, (float*)d_out);
}
